// LinearAttention_49847390437475
// MI455X (gfx1250) — compile-verified, matrix operand order checked by reading, not run on hardware
//
#include <hip/hip_runtime.h>
#include <hip/hip_bf16.h>

typedef __attribute__((ext_vector_type(16))) _Float16 v16h;
typedef __attribute__((ext_vector_type(8)))  _Float16 v8h;
typedef __attribute__((ext_vector_type(2)))  _Float16 v2h;
typedef __attribute__((ext_vector_type(8)))  float    v8f;
typedef __attribute__((ext_vector_type(4)))  unsigned int v4u;
typedef __attribute__((ext_vector_type(8)))  int      v8i;
typedef __attribute__((ext_vector_type(4)))  int      v4i;

#define WMMA_F16(a, b, c) __builtin_amdgcn_wmma_f32_16x16x32_f16(false, (a), false, (b), (short)0, (c), false, false)

static constexpr int B_  = 4;
static constexpr int N_  = 4096;
static constexpr int C_  = 1024;
static constexpr int H_  = 8;
static constexpr int D_  = 128;
static constexpr int M_  = B_ * N_;      // 16384
static constexpr int NQK = 2 * C_;       // 2048
static constexpr int KL  = 3 * C_;       // 3072
static constexpr int KCH = 8;            // k-split chunks for kv GEMM
static constexpr int NCH = 16;           // n-split chunks for k_mean

// ---------------------------------------------------------------------------
// TDM: DMA a B tile (128 cols x 32 halves, col stride = src row length) into
// LDS with hardware padding: 64B of data then 32B pad -> 48-half LDS row
// stride (bank-conflict padding done by the Tensor Data Mover itself).
// D# per CDNA5 ISA 8.3-8.6. Issued by one wave; tracked by TENSORcnt.
// 6-arg builtin on this toolchain: (g0 u32x4, g1 i32x8, i32x4, i32x4, i32x8, cpol).
// ---------------------------------------------------------------------------
__device__ __forceinline__ void tdm_load_b_tile(const _Float16* gsrc, _Float16* lds_dst,
                                                unsigned col_stride_halves) {
    unsigned long long ga = (unsigned long long)(size_t)gsrc;
    v4u g0;
    g0[0] = 1u;                                              // count=1, user mode, no gather
    g0[1] = (unsigned)(size_t)lds_dst;                       // lds_addr = addr[31:0]
    g0[2] = (unsigned)ga;                                    // global_addr[31:0]
    g0[3] = ((unsigned)(ga >> 32) & 0x01FFFFFFu) | (2u << 30); // global_addr[56:32] | type=2
    v8i g1;
    g1[0] = (1 << 16)        // data_size = 1 -> 2 bytes
          | (1 << 20)        // pad_enable
          | (3 << 22)        // pad_interval = 16 DWORDs (64B) between pads
          | (7 << 25);       // pad_amount   = 8 DWORDs (32B)
    g1[1] = 0;                                               // atomic barrier off; tensor_dim0 lo16 = 0
    g1[2] = (int)(((1u << 20) >> 16));                       // tensor_dim0 = 1<<20 (no clip)
    g1[3] = (int)(((1u << 20) >> 16) | (32u << 16));         // tensor_dim1 = 1<<20 | tile_dim0 = 32
    g1[4] = 128;                                             // tile_dim1 = 128, tile_dim2 = 0
    g1[5] = (int)col_stride_halves;                          // tensor_dim0_stride lo32
    g1[6] = 0;
    g1[7] = 0;
    v4i gz4 = {0, 0, 0, 0};
    v8i gz8 = {0, 0, 0, 0, 0, 0, 0, 0};
    __builtin_amdgcn_tensor_load_to_lds(g0, g1, gz4, gz4, gz8, 0);
}

// ---------------------------------------------------------------------------
// prep_x: x f32 -> xh [m,c] f16 (row major) and xT [b,c,n] f16 via LDS transpose
// ---------------------------------------------------------------------------
__global__ void la_prep_x(const float* __restrict__ x, _Float16* __restrict__ xh,
                          _Float16* __restrict__ xT) {
    __shared__ __align__(32) _Float16 tile[64][80];
    const int tid = threadIdx.x;
    const int ch0 = blockIdx.x * 64;
    const int n0  = blockIdx.y * 64;
    const int b   = blockIdx.z;

#pragma unroll
    for (int rr = 0; rr < 16; ++rr) {
        int idx = rr * 256 + tid;
        int nl  = idx >> 6;
        int cl  = idx & 63;
        long gi = ((long)(b * N_ + n0 + nl)) * C_ + ch0 + cl;
        _Float16 v = (_Float16)x[gi];
        xh[gi] = v;
        tile[cl][nl] = v;
    }
    __syncthreads();
    {
        int cl  = tid >> 2;
        int seg = tid & 3;
        v16h pack = *(const v16h*)&tile[cl][seg * 16];
        *(v16h*)(xT + ((long)(b * C_ + ch0 + cl)) * N_ + n0 + seg * 16) = pack;
    }
}

__global__ void la_prep_qkw(const float* __restrict__ w, _Float16* __restrict__ wT) {
    long idx = (long)blockIdx.x * blockDim.x + threadIdx.x;
    if (idx >= (long)C_ * NQK) return;
    int k = (int)(idx / NQK);
    int j = (int)(idx % NQK);
    wT[(long)j * C_ + k] = (_Float16)w[idx];
}

__global__ void la_prep_lepew(const float* __restrict__ w, _Float16* __restrict__ wT) {
    long idx = (long)blockIdx.x * blockDim.x + threadIdx.x;
    if (idx >= (long)3 * C_ * C_) return;
    int t = (int)(idx >> 20);
    int k = (int)((idx >> 10) & (C_ - 1));
    int j = (int)(idx & (C_ - 1));
    wT[(long)j * KL + t * C_ + k] = (_Float16)w[idx];
}

// ---------------------------------------------------------------------------
// GEMM 1: qk = elu(x @ qk_w + b) + 1 -> q_act, k_act (f16)
// 8 waves, tile 128x128, K=1024. B tile double-buffered in LDS via TDM.
// ---------------------------------------------------------------------------
__global__ void __launch_bounds__(256, 1)
la_gemm_qk(const _Float16* __restrict__ xh, const _Float16* __restrict__ wT,
           const float* __restrict__ bias,
           _Float16* __restrict__ q_act, _Float16* __restrict__ k_act) {
    __shared__ __align__(32) _Float16 lb[2][128][48];
    const int tid  = threadIdx.x;
    const int wave = tid >> 5, lane = tid & 31;
    const int lm = lane & 15, lh = lane >> 4;
    const int row0 = blockIdx.y * 128 + wave * 16;
    const int col0 = blockIdx.x * 128;

    v8f zero = {};
    v8f acc[8];
#pragma unroll
    for (int j = 0; j < 8; ++j) acc[j] = zero;

    const _Float16* Arow = xh + (long)(row0 + lm) * C_ + lh * 16;
    const _Float16* Bsrc = wT + (long)col0 * C_;        // tile origin, k offset added per step

    if (tid < 32) tdm_load_b_tile(Bsrc, &lb[0][0][0], C_);
    __builtin_amdgcn_s_wait_tensorcnt(0);               // no-op for non-issuing waves
    __syncthreads();

    const int NT = C_ / 32;
    for (int kt = 0; kt < NT; ++kt) {
        if (tid < 32 && kt + 1 < NT)
            tdm_load_b_tile(Bsrc + (kt + 1) * 32, &lb[(kt + 1) & 1][0][0], C_);
        v16h a = *(const v16h*)(Arow + kt * 32);
#pragma unroll
        for (int j = 0; j < 8; ++j) {
            v16h bf = *(const v16h*)&lb[kt & 1][j * 16 + lm][lh * 16];
            acc[j] = WMMA_F16(a, bf, acc[j]);
        }
        __builtin_amdgcn_s_wait_tensorcnt(0);
        __syncthreads();
    }

#pragma unroll
    for (int j = 0; j < 8; ++j) {
        int col = col0 + j * 16 + lm;
        float bv = bias[col];
#pragma unroll
        for (int r = 0; r < 8; ++r) {
            int row = row0 + r + 8 * lh;
            float v = acc[j][r] + bv;
            v = (v > 0.0f) ? (v + 1.0f) : __expf(v);   // elu(v)+1
            if (col < C_) q_act[(long)row * C_ + col] = (_Float16)v;
            else          k_act[(long)row * C_ + (col - C_)] = (_Float16)v;
        }
    }
}

// ---------------------------------------------------------------------------
// k_mean partials over n-chunks, then reduce
// ---------------------------------------------------------------------------
__global__ void la_kmean_part(const _Float16* __restrict__ k_act, float* __restrict__ km_part) {
    int b = blockIdx.y, chunk = blockIdx.z;
    int ch = blockIdx.x * blockDim.x + threadIdx.x;
    float s = 0.0f;
    for (int n2 = 0; n2 < N_ / NCH; ++n2) {
        int n = chunk * (N_ / NCH) + n2;
        s += (float)k_act[(long)(b * N_ + n) * C_ + ch];
    }
    km_part[((long)chunk * B_ + b) * C_ + ch] = s;
}

__global__ void la_kmean_red(const float* __restrict__ km_part, float* __restrict__ k_mean) {
    int idx = blockIdx.x * blockDim.x + threadIdx.x;
    if (idx >= B_ * C_) return;
    int b = idx >> 10, ch = idx & (C_ - 1);
    float s = 0.0f;
#pragma unroll
    for (int c2 = 0; c2 < NCH; ++c2) s += km_part[((long)c2 * B_ + b) * C_ + ch];
    k_mean[idx] = s * (1.0f / (float)N_);
}

// ---------------------------------------------------------------------------
// RoPE: qr [m,c] f16 row-major; krT [b,c,n] f16 via LDS transpose.
// ---------------------------------------------------------------------------
__global__ void la_rope(const _Float16* __restrict__ q_act, const _Float16* __restrict__ k_act,
                        _Float16* __restrict__ qr, _Float16* __restrict__ krT) {
    __shared__ __align__(32) _Float16 tile[64][80];
    const int tid = threadIdx.x;
    const int i0  = blockIdx.x * 32;
    const int n0  = blockIdx.y * 64;
    const int b   = blockIdx.z;

    const int il = tid & 31;
    const int i  = i0 + il;
    const float theta = __expf(-(float)i * (9.210340371976184f / 512.0f));

#pragma unroll
    for (int rr = 0; rr < 8; ++rr) {
        int nl = (tid >> 5) * 8 + rr;
        int n  = n0 + nl;
        long m = (long)b * N_ + n;
        float s, c;
        __sincosf((float)n * theta, &s, &c);

        v2h qp = *(const v2h*)(q_act + m * C_ + 2 * i);
        v2h kp = *(const v2h*)(k_act + m * C_ + 2 * i);
        float q0 = (float)qp[0], q1 = (float)qp[1];
        float k0 = (float)kp[0], k1 = (float)kp[1];

        qr[m * C_ + i]       = (_Float16)(c * q0 - s * q1);
        qr[m * C_ + 512 + i] = (_Float16)(c * q1 + s * q0);
        tile[il][nl]      = (_Float16)(c * k0 - s * k1);
        tile[32 + il][nl] = (_Float16)(c * k1 + s * k0);
    }
    __syncthreads();
    {
        int cl  = tid >> 2;
        int seg = tid & 3;
        int ch  = (cl < 32) ? (i0 + cl) : (512 + i0 + (cl - 32));
        v16h pack = *(const v16h*)&tile[cl][seg * 16];
        *(v16h*)(krT + ((long)(b * C_ + ch)) * N_ + n0 + seg * 16) = pack;
    }
}

// ---------------------------------------------------------------------------
// z[m,h] = 1 / (dot(q_act[m, h*128:], k_mean[b, h*128:]) + 1e-6)
// ---------------------------------------------------------------------------
__global__ void la_z(const _Float16* __restrict__ q_act, const float* __restrict__ k_mean,
                     float* __restrict__ z) {
    long idx = (long)blockIdx.x * blockDim.x + threadIdx.x;
    if (idx >= (long)M_ * H_) return;
    int m = (int)(idx >> 3);
    int h = (int)(idx & 7);
    int b = m >> 12;
    const _Float16* qp = q_act + (long)m * C_ + h * D_;
    const float* kp = k_mean + b * C_ + h * D_;
    float s = 0.0f;
#pragma unroll 4
    for (int d = 0; d < D_; ++d) s += (float)qp[d] * kp[d];
    z[idx] = 1.0f / (s + 1e-6f);
}

// ---------------------------------------------------------------------------
// kv GEMM (k-split): partial kv[d,e] over a 512-length n-chunk.
// Grid (B*H, KCH). B tile double-buffered in LDS via TDM.
// ---------------------------------------------------------------------------
__global__ void __launch_bounds__(256, 1)
la_gemm_kv(const _Float16* __restrict__ krT, const _Float16* __restrict__ xT,
           float* __restrict__ kv_part) {
    __shared__ __align__(32) _Float16 lb[2][128][48];
    const int bh = blockIdx.x;
    const int b = bh >> 3, h = bh & 7;
    const int kbase = blockIdx.y * (N_ / KCH);
    const int tid  = threadIdx.x;
    const int wave = tid >> 5, lane = tid & 31;
    const int lm = lane & 15, lh = lane >> 4;

    v8f zero = {};
    v8f acc[8];
#pragma unroll
    for (int j = 0; j < 8; ++j) acc[j] = zero;

    const _Float16* Arow = krT + (long)(b * C_ + h * D_ + wave * 16 + lm) * N_ + kbase + lh * 16;
    const _Float16* Bsrc = xT  + (long)(b * C_ + h * D_) * N_ + kbase;

    if (tid < 32) tdm_load_b_tile(Bsrc, &lb[0][0][0], N_);
    __builtin_amdgcn_s_wait_tensorcnt(0);
    __syncthreads();

    const int NT = (N_ / KCH) / 32;     // 16
    for (int kt = 0; kt < NT; ++kt) {
        if (tid < 32 && kt + 1 < NT)
            tdm_load_b_tile(Bsrc + (kt + 1) * 32, &lb[(kt + 1) & 1][0][0], N_);
        v16h a = *(const v16h*)(Arow + kt * 32);
#pragma unroll
        for (int j = 0; j < 8; ++j) {
            v16h bf = *(const v16h*)&lb[kt & 1][j * 16 + lm][lh * 16];
            acc[j] = WMMA_F16(a, bf, acc[j]);
        }
        __builtin_amdgcn_s_wait_tensorcnt(0);
        __syncthreads();
    }

    float* dst = kv_part + ((long)(blockIdx.y * (B_ * H_) + bh)) * D_ * D_;
#pragma unroll
    for (int j = 0; j < 8; ++j) {
        int e = j * 16 + lm;
#pragma unroll
        for (int r = 0; r < 8; ++r) {
            int d = wave * 16 + r + 8 * lh;
            dst[(long)d * D_ + e] = acc[j][r];
        }
    }
}

__global__ void la_kv_red(const float* __restrict__ kv_part, _Float16* __restrict__ kvT) {
    int idx = blockIdx.x * blockDim.x + threadIdx.x;
    if (idx >= B_ * H_ * D_ * D_) return;
    int bh = idx >> 14;
    int d  = (idx >> 7) & 127;
    int e  = idx & 127;
    float s = 0.0f;
#pragma unroll
    for (int kc = 0; kc < KCH; ++kc)
        s += kv_part[((long)(kc * (B_ * H_) + bh)) * D_ * D_ + d * D_ + e];
    kvT[((long)(bh * D_ + e)) * D_ + d] = (_Float16)(s * (1.0f / 64.0f));
}

// ---------------------------------------------------------------------------
// lepe GEMM (implicit im2col conv1d k=3, SAME) -> out = conv + bias (f32)
// Grid (C/128, M/128). K=3072. B tile double-buffered in LDS via TDM.
// ---------------------------------------------------------------------------
__global__ void __launch_bounds__(256, 1)
la_gemm_lepe(const _Float16* __restrict__ xh, const _Float16* __restrict__ wT,
             const float* __restrict__ bias, float* __restrict__ out) {
    __shared__ __align__(32) _Float16 lb[2][128][48];
    const int tid  = threadIdx.x;
    const int wave = tid >> 5, lane = tid & 31;
    const int lm = lane & 15, lh = lane >> 4;
    const int row0 = blockIdx.y * 128 + wave * 16;
    const int col0 = blockIdx.x * 128;

    const int m = row0 + lm;
    const int b = m >> 12;
    const int n = m & (N_ - 1);

    v8f zero = {};
    v16h azero = {};
    v8f acc[8];
#pragma unroll
    for (int j = 0; j < 8; ++j) acc[j] = zero;

    const _Float16* Bsrc = wT + (long)col0 * KL;

    if (tid < 32) tdm_load_b_tile(Bsrc, &lb[0][0][0], KL);
    __builtin_amdgcn_s_wait_tensorcnt(0);
    __syncthreads();

    const int NT = KL / 32;             // 96
    for (int kt = 0; kt < NT; ++kt) {
        if (tid < 32 && kt + 1 < NT)
            tdm_load_b_tile(Bsrc + (kt + 1) * 32, &lb[(kt + 1) & 1][0][0], KL);
        int kglob = kt * 32;
        int t  = kglob >> 10;
        int kk = kglob & (C_ - 1);
        int nn = n + t - 1;
        v16h a = azero;
        if ((unsigned)nn < (unsigned)N_)
            a = *(const v16h*)(xh + ((long)b * N_ + nn) * C_ + kk + lh * 16);
#pragma unroll
        for (int j = 0; j < 8; ++j) {
            v16h bf = *(const v16h*)&lb[kt & 1][j * 16 + lm][lh * 16];
            acc[j] = WMMA_F16(a, bf, acc[j]);
        }
        __builtin_amdgcn_s_wait_tensorcnt(0);
        __syncthreads();
    }

#pragma unroll
    for (int j = 0; j < 8; ++j) {
        int col = col0 + j * 16 + lm;
        float bv = bias[col];
#pragma unroll
        for (int r = 0; r < 8; ++r) {
            int row = row0 + r + 8 * lh;
            out[(long)row * C_ + col] = acc[j][r] + bv;
        }
    }
}

// ---------------------------------------------------------------------------
// out GEMM: out[m, h*128+e] += z[m,h] * (qr[m, h*128:] . kvT[(b*H+h)*128+e, :])
// ---------------------------------------------------------------------------
__global__ void __launch_bounds__(256, 1)
la_gemm_out(const _Float16* __restrict__ qr, const _Float16* __restrict__ kvT,
            const float* __restrict__ z, float* __restrict__ out) {
    const int h = blockIdx.x;
    const int wave = threadIdx.x >> 5;
    const int lane = threadIdx.x & 31;
    const int lm = lane & 15, lh = lane >> 4;
    const int row0 = blockIdx.y * 128 + wave * 16;
    const int b = (blockIdx.y * 128) >> 12;
    const int bh = b * H_ + h;

    v8f zero = {};
    v8f acc[8];
#pragma unroll
    for (int j = 0; j < 8; ++j) acc[j] = zero;

    const _Float16* Arow  = qr  + (long)(row0 + lm) * C_ + h * D_ + lh * 16;
    const _Float16* Bbase = kvT + (long)(bh * D_ + lm) * D_ + lh * 16;

#pragma unroll
    for (int k0 = 0; k0 < D_; k0 += 32) {
        v16h a = *(const v16h*)(Arow + k0);
#pragma unroll
        for (int j = 0; j < 8; ++j) {
            v16h bf = *(const v16h*)(Bbase + (long)j * 16 * D_ + k0);
            acc[j] = WMMA_F16(a, bf, acc[j]);
        }
    }

#pragma unroll
    for (int r = 0; r < 8; ++r) {
        int row = row0 + r + 8 * lh;
        float zv = z[(long)row * H_ + h];
#pragma unroll
        for (int j = 0; j < 8; ++j) {
            int e = j * 16 + lm;
            long idx = (long)row * C_ + h * D_ + e;
            out[idx] += acc[j][r] * zv;
        }
    }
}

// ---------------------------------------------------------------------------
// Host launch
// ---------------------------------------------------------------------------
extern "C" void kernel_launch(void* const* d_in, const int* in_sizes, int n_in,
                              void* d_out, int out_size, void* d_ws, size_t ws_size,
                              hipStream_t stream) {
    const float* x      = (const float*)d_in[0];
    const float* qk_w   = (const float*)d_in[1];
    const float* qk_b   = (const float*)d_in[2];
    const float* lepe_w = (const float*)d_in[3];
    const float* lepe_b = (const float*)d_in[4];
    float* out = (float*)d_out;

    char* ws = (char*)d_ws;
    size_t off = 0;
    auto alloc = [&](size_t bytes) { void* p = ws + off; off = (off + bytes + 255) & ~(size_t)255; return p; };

    _Float16* xh      = (_Float16*)alloc((size_t)M_ * C_ * 2);
    _Float16* xT      = (_Float16*)alloc((size_t)M_ * C_ * 2);
    _Float16* qk_wT   = (_Float16*)alloc((size_t)NQK * C_ * 2);
    _Float16* lepe_wT = (_Float16*)alloc((size_t)C_ * KL * 2);
    _Float16* q_act   = (_Float16*)alloc((size_t)M_ * C_ * 2);
    _Float16* k_act   = (_Float16*)alloc((size_t)M_ * C_ * 2);
    _Float16* qr      = (_Float16*)alloc((size_t)M_ * C_ * 2);
    _Float16* krT     = (_Float16*)alloc((size_t)M_ * C_ * 2);
    _Float16* kvT     = (_Float16*)alloc((size_t)B_ * H_ * D_ * D_ * 2);
    float*    kv_part = (float*)alloc((size_t)KCH * B_ * H_ * D_ * D_ * 4);
    float*    k_mean  = (float*)alloc((size_t)B_ * C_ * 4);
    float*    km_part = (float*)alloc((size_t)NCH * B_ * C_ * 4);
    float*    zbuf    = (float*)alloc((size_t)M_ * H_ * 4);

    const int T = 256;

    la_prep_x<<<dim3(C_ / 64, N_ / 64, B_), T, 0, stream>>>(x, xh, xT);
    la_prep_qkw<<<dim3(((long)C_ * NQK + T - 1) / T), T, 0, stream>>>(qk_w, qk_wT);
    la_prep_lepew<<<dim3(((long)3 * C_ * C_ + T - 1) / T), T, 0, stream>>>(lepe_w, lepe_wT);

    la_gemm_qk<<<dim3(NQK / 128, M_ / 128), T, 0, stream>>>(xh, qk_wT, qk_b, q_act, k_act);

    la_kmean_part<<<dim3(C_ / T, B_, NCH), T, 0, stream>>>(k_act, km_part);
    la_kmean_red<<<dim3((B_ * C_ + T - 1) / T), T, 0, stream>>>(km_part, k_mean);
    la_rope<<<dim3(512 / 32, N_ / 64, B_), T, 0, stream>>>(q_act, k_act, qr, krT);
    la_z<<<dim3(((long)M_ * H_ + T - 1) / T), T, 0, stream>>>(q_act, k_mean, zbuf);

    la_gemm_kv<<<dim3(B_ * H_, KCH), T, 0, stream>>>(krT, xT, kv_part);
    la_kv_red<<<dim3((B_ * H_ * D_ * D_ + T - 1) / T), T, 0, stream>>>(kv_part, kvT);

    la_gemm_lepe<<<dim3(C_ / 128, M_ / 128), T, 0, stream>>>(xh, lepe_wT, lepe_b, out);
    la_gemm_out<<<dim3(H_, M_ / 128), T, 0, stream>>>(qr, kvT, zbuf, out);

    (void)in_sizes; (void)n_in; (void)out_size; (void)ws_size;
}
